// MagnitudePruningModel_15418932592954
// MI455X (gfx1250) — matrix operand order repaired and checked by reading, compiled, not run on hardware
//
#include <hip/hip_runtime.h>
#include <hip/hip_bf16.h>
#include <math.h>

typedef __attribute__((ext_vector_type(16))) __bf16        v16bf;
typedef __attribute__((ext_vector_type(8)))  float         v8f;
typedef __attribute__((ext_vector_type(4)))  unsigned int  u32x4;

union FragBF { v16bf v; u32x4 q[2]; unsigned short u[16]; };

__device__ __forceinline__ unsigned short f2bf(float f) {
    unsigned int u = __float_as_uint(f);
    u += 0x7FFFu + ((u >> 16) & 1u);          // round-to-nearest-even
    return (unsigned short)(u >> 16);
}

__device__ __forceinline__ float gelu_tanh(float v) {
    const float c = 0.7978845608028654f;
    float t = tanhf(c * (v + 0.044715f * v * v * v));
    return 0.5f * v * (1.0f + t);
}

// ---------------- conversion kernels ----------------

__global__ void k_f32_to_bf16(const float* __restrict__ src,
                              unsigned short* __restrict__ dst, int n) {
    int i = blockIdx.x * blockDim.x + threadIdx.x;
    if (i < n) dst[i] = f2bf(src[i]);
}

// src is K x N row-major (fp32); dst is N x K row-major (bf16)  => dst = src^T
__global__ void k_f32_to_bf16_T(const float* __restrict__ src,
                                unsigned short* __restrict__ dst, int K, int N) {
    int i = blockIdx.x * blockDim.x + threadIdx.x;
    if (i < K * N) {
        int k = i / N, n = i - k * N;
        dst[(size_t)n * K + k] = f2bf(src[i]);
    }
}

// ---------------- layernorm (fp32 in -> bf16 out), one row / block ----------------

__global__ void k_layernorm_bf16(const float* __restrict__ x,
                                 const float* __restrict__ g,
                                 const float* __restrict__ b,
                                 unsigned short* __restrict__ out, int D) {
    __shared__ float red[256];
    int row = blockIdx.x, tid = threadIdx.x;
    const float* xr = x + (size_t)row * D;
    float vals[4];
    float s = 0.f;
    for (int j = 0; j < 4; ++j) { vals[j] = xr[tid * 4 + j]; s += vals[j]; }
    red[tid] = s; __syncthreads();
    for (int st = 128; st > 0; st >>= 1) { if (tid < st) red[tid] += red[tid + st]; __syncthreads(); }
    float mu = red[0] / D;
    __syncthreads();
    float vs = 0.f;
    for (int j = 0; j < 4; ++j) { float d = vals[j] - mu; vs += d * d; }
    red[tid] = vs; __syncthreads();
    for (int st = 128; st > 0; st >>= 1) { if (tid < st) red[tid] += red[tid + st]; __syncthreads(); }
    float inv = rsqrtf(red[0] / D + 1e-5f);
    unsigned short* orow = out + (size_t)row * D;
    for (int j = 0; j < 4; ++j) {
        int c = tid * 4 + j;
        orow[c] = f2bf((vals[j] - mu) * inv * g[c] + b[c]);
    }
}

// ---------------- magnitude prune: exact top-K threshold per row ----------------
// Binary search on |x| bit patterns (monotone as uint for non-negative floats).
__global__ void k_prune(float* __restrict__ x, int D, int Keep) {
    __shared__ unsigned int cnt;
    int row = blockIdx.x, tid = threadIdx.x;
    float* xr = x + (size_t)row * D;
    unsigned int a[4];
    for (int j = 0; j < 4; ++j) a[j] = __float_as_uint(fabsf(xr[tid * 4 + j]));
    unsigned int lo = 0;
    for (int bit = 31; bit >= 0; --bit) {
        unsigned int t = lo | (1u << bit);
        if (tid == 0) cnt = 0;
        __syncthreads();
        unsigned int c = 0;
        for (int j = 0; j < 4; ++j) c += (a[j] >= t) ? 1u : 0u;
        atomicAdd(&cnt, c);
        __syncthreads();
        if (cnt >= (unsigned int)Keep) lo = t;   // thr is largest t with count >= K
        __syncthreads();
    }
    for (int j = 0; j < 4; ++j)
        if (a[j] < lo) xr[tid * 4 + j] = 0.f;
}

// ---------------- fused bf16 WMMA GEMM ----------------
// A:  M x K bf16 row-major
// Bt: N x K bf16 row-major (pre-transposed weights)
// out = act(A @ B + bias) [+ residual]; writes fp32 and/or bf16
// Tiling: block = 8 wave32s; wave tile 32x64 (2 A-frags x 4 B-tiles, 8 v8f acc);
// block tile 256x64. B tile double-buffered in LDS (2 x 4KB), one barrier/k-step.
__global__ __launch_bounds__(256) void k_gemm_bf16(
    const unsigned short* __restrict__ A,
    const unsigned short* __restrict__ Bt,
    const float* __restrict__ bias,
    const float* __restrict__ residual,
    float* __restrict__ outF,
    unsigned short* __restrict__ outB,
    int M, int N, int K, int doGelu)
{
    __shared__ __align__(16) unsigned short sB[2][64 * 32];

    int tid  = threadIdx.x;
    int wave = tid >> 5, lane = tid & 31;
    int half = lane >> 4, r = lane & 15;
    int n0 = blockIdx.x * 64;
    int m0 = blockIdx.y * 256 + wave * 32;   // 8 waves x 32 rows = 256 rows / block

    v8f acc0[4], acc1[4];
    for (int t = 0; t < 4; ++t)
        for (int j = 0; j < 8; ++j) { acc0[t][j] = 0.0f; acc1[t][j] = 0.0f; }

    int srow = tid >> 2, schunk = tid & 3;   // 256 threads stage 64 rows x 64B
    const unsigned short* bsrc  = Bt + (size_t)(n0 + srow) * K + schunk * 8;
    const unsigned short* arow0 = A + (size_t)(m0 + r) * K;        // frag 0: rows m0..m0+15
    const unsigned short* arow1 = A + (size_t)(m0 + 16 + r) * K;   // frag 1: rows m0+16..m0+31

    // prologue: stage k-step 0 into buffer 0
    ((u32x4*)sB[0])[srow * 4 + schunk] = *(const u32x4*)(bsrc);
    __syncthreads();

    int nsteps = K >> 5;
    for (int s = 0; s < nsteps; ++s) {
        int buf = s & 1;
        int k0  = s << 5;

        if (s + 1 < nsteps) {
            ((u32x4*)sB[buf ^ 1])[srow * 4 + schunk] = *(const u32x4*)(bsrc + k0 + 32);
            if (s + 2 < nsteps) __builtin_prefetch(bsrc + k0 + 64, 0, 1);
        }

        // 16-bit A fragment layout (wave32): lanes 0-15 hold K{0..7,16..23},
        // lanes 16-31 hold K{8..15,24..31}, row M = lane%16.
        FragBF a0, a1;
        a0.q[0] = *(const u32x4*)(arow0 + k0 + half * 8);
        a0.q[1] = *(const u32x4*)(arow0 + k0 + 16 + half * 8);
        a1.q[0] = *(const u32x4*)(arow1 + k0 + half * 8);
        a1.q[1] = *(const u32x4*)(arow1 + k0 + 16 + half * 8);

#pragma unroll
        for (int t = 0; t < 4; ++t) {
            // B fragment: lane holds column c = lane%16 of tile t, K = half*8..+7, 16+half*8..+7
            FragBF bfrag;
            const u32x4* bp = (const u32x4*)(sB[buf] + (t * 16 + r) * 32);
            bfrag.q[0] = bp[half];
            bfrag.q[1] = bp[2 + half];
            acc0[t] = __builtin_amdgcn_wmma_f32_16x16x32_bf16(
                false, a0.v, false, bfrag.v, (short)0, acc0[t], false, false);
            acc1[t] = __builtin_amdgcn_wmma_f32_16x16x32_bf16(
                false, a1.v, false, bfrag.v, (short)0, acc1[t], false, false);
        }
        __syncthreads();
    }

    // Epilogue: D layout — lane (half,r): acc[t][j] = D[M = half*8 + j][N = r]
#pragma unroll
    for (int t = 0; t < 4; ++t) {
        int col = n0 + t * 16 + r;
        float bb = bias ? bias[col] : 0.0f;
#pragma unroll
        for (int j = 0; j < 8; ++j) {
            int rowm = m0 + half * 8 + j;
            {
                float v = acc0[t][j] + bb;
                if (doGelu) v = gelu_tanh(v);
                size_t idx = (size_t)rowm * N + col;
                if (residual) v += residual[idx];
                if (outF) outF[idx] = v;
                if (outB) outB[idx] = f2bf(v);
            }
            {
                float v = acc1[t][j] + bb;
                if (doGelu) v = gelu_tanh(v);
                size_t idx = (size_t)(rowm + 16) * N + col;
                if (residual) v += residual[idx];
                if (outF) outF[idx] = v;
                if (outB) outB[idx] = f2bf(v);
            }
        }
    }
}

// ---------------- launch ----------------

extern "C" void kernel_launch(void* const* d_in, const int* in_sizes, int n_in,
                              void* d_out, int out_size, void* d_ws, size_t ws_size,
                              hipStream_t stream) {
    (void)in_sizes; (void)n_in; (void)out_size; (void)ws_size;
    const float* frames = (const float*)d_in[0];
    const float* W_in   = (const float*)d_in[1];
    const float* b_in   = (const float*)d_in[2];
    const float* ln_g   = (const float*)d_in[3];
    const float* ln_b   = (const float*)d_in[4];
    const float* W1     = (const float*)d_in[5];
    const float* b1     = (const float*)d_in[6];
    const float* W2     = (const float*)d_in[7];
    const float* b2     = (const float*)d_in[8];
    const float* fn_g   = (const float*)d_in[9];
    const float* fn_b   = (const float*)d_in[10];
    const float* W_head = (const float*)d_in[11];
    const float* b_head = (const float*)d_in[12];

    const int T = 4 * 4096;          // 16384 tokens
    const int F = 128, D = 1024, DFF = 4096, V = 512, L = 4;

    char* ws = (char*)d_ws;
    size_t off = 0;
    auto carve = [&](size_t bytes) -> char* {
        char* p = ws + off; off += (bytes + 255) & ~(size_t)255; return p;
    };
    unsigned short* WtIn = (unsigned short*)carve((size_t)D * F * 2);
    unsigned short* Wt1  = (unsigned short*)carve((size_t)L * DFF * D * 2);
    unsigned short* Wt2  = (unsigned short*)carve((size_t)L * D * DFF * 2);
    unsigned short* WtHd = (unsigned short*)carve((size_t)V * D * 2);
    unsigned short* fbf  = (unsigned short*)carve((size_t)T * F * 2);
    float*          x    = (float*)carve((size_t)T * D * 4);
    unsigned short* hn   = (unsigned short*)carve((size_t)T * D * 2);
    unsigned short* h    = (unsigned short*)carve((size_t)T * DFF * 2);

    // one-time-per-launch conversions (bf16 + weight transpose)
    {
        int n = T * F;
        k_f32_to_bf16<<<(n + 255) / 256, 256, 0, stream>>>(frames, fbf, n);
        k_f32_to_bf16_T<<<(F * D + 255) / 256, 256, 0, stream>>>(W_in, WtIn, F, D);
        for (int l = 0; l < L; ++l) {
            k_f32_to_bf16_T<<<(D * DFF + 255) / 256, 256, 0, stream>>>(
                W1 + (size_t)l * D * DFF, Wt1 + (size_t)l * DFF * D, D, DFF);
            k_f32_to_bf16_T<<<(DFF * D + 255) / 256, 256, 0, stream>>>(
                W2 + (size_t)l * DFF * D, Wt2 + (size_t)l * D * DFF, DFF, D);
        }
        k_f32_to_bf16_T<<<(D * V + 255) / 256, 256, 0, stream>>>(W_head, WtHd, D, V);
    }

    dim3 blk(256);
    // x = frames @ W_in + b_in
    k_gemm_bf16<<<dim3(D / 64, T / 256), blk, 0, stream>>>(
        fbf, WtIn, b_in, nullptr, x, nullptr, T, D, F, 0);

    for (int l = 0; l < L; ++l) {
        k_layernorm_bf16<<<T, 256, 0, stream>>>(x, ln_g + l * D, ln_b + l * D, hn, D);
        // h = gelu(hn @ W1 + b1)  (bf16 out)
        k_gemm_bf16<<<dim3(DFF / 64, T / 256), blk, 0, stream>>>(
            hn, Wt1 + (size_t)l * DFF * D, b1 + l * DFF, nullptr, nullptr, h, T, DFF, D, 1);
        // x = x + h @ W2 + b2  (fp32, in place residual)
        k_gemm_bf16<<<dim3(D / 64, T / 256), blk, 0, stream>>>(
            h, Wt2 + (size_t)l * D * DFF, b2 + l * D, x, x, nullptr, T, D, DFF, 0);
        k_prune<<<T, 256, 0, stream>>>(x, D, D / 2);
    }

    k_layernorm_bf16<<<T, 256, 0, stream>>>(x, fn_g, fn_b, hn, D);
    // out = hn @ W_head + b_head
    k_gemm_bf16<<<dim3(V / 64, T / 256), blk, 0, stream>>>(
        hn, WtHd, b_head, nullptr, (float*)d_out, nullptr, T, V, D, 0);
}
